// Graph_76424648066041
// MI455X (gfx1250) — matrix operand order repaired and checked by reading, not compiled, not run on hardware
//
#include <hip/hip_runtime.h>
#include <hip/hip_bf16.h>
#include <math.h>

// ---------------------------------------------------------------------------
// Types for CDNA5 WMMA (wave32, 16x16x32 bf16 -> f32)
// ---------------------------------------------------------------------------
typedef __attribute__((ext_vector_type(16))) __bf16 v16bf;
typedef __attribute__((ext_vector_type(8)))  float  v8f;

union Frag {            // one WMMA A/B operand: 16 bf16 = 32 bytes = 2 x uint4
    uint4 u[2];
    v16bf v;
};

__device__ __forceinline__ unsigned short f2bf(float f) {
    // round-to-nearest-even f32 -> bf16
    unsigned u = __float_as_uint(f);
    u += 0x7FFFu + ((u >> 16) & 1u);
    return (unsigned short)(u >> 16);
}

// ---------------------------------------------------------------------------
// Tiled batched GEMM:  C[b] = act( A[b] (MxK bf16) * B[b] (KxN bf16) [+C][+bias] )
// Block tile 128x128, K-step 32, 256 threads = 8 waves, each wave 32x64.
// Double-buffered LDS: one barrier per K-step; WMMAs overlap global loads.
// ACT: 0=none 1=relu 2=gelu(erf) 3=sigmoid. Compile-time epilogue flags.
// Assumes M%128==0, N%128==0, K%32==0 (true for all shapes here).
// ---------------------------------------------------------------------------
#define LPA 40   // LDS pitch (bf16 elems) for A tile rows (128 x 32)
#define LPB 40   // LDS pitch (bf16 elems) for transposed B tile rows (128 x 32)

template<int ACT, bool ACC, bool WF, bool WB, bool BIAS>
__global__ __launch_bounds__(256)
void k_gemm(const unsigned short* __restrict__ A,
            const unsigned short* __restrict__ B,
            float*          __restrict__ Cf,
            unsigned short* __restrict__ Cb,
            const float*    __restrict__ bias,
            int M, int N, int K,
            long long sA, long long sB, long long sC)
{
    __shared__ unsigned short lA[2][128 * LPA];   // 2 x 10 KB
    __shared__ unsigned short lB[2][128 * LPB];   // 2 x 10 KB (transposed [n][k])

    const int tid  = threadIdx.x;
    const int wave = tid >> 5;
    const int lane = tid & 31;
    const int lh   = lane >> 4;        // lane-half (0/1), selects K sub-block
    const int l16  = lane & 15;
    const int wm   = wave & 3;         // M quadrant (32 rows each)
    const int wn   = wave >> 2;        // N half    (64 cols each)

    // global->LDS copy mapping
    const int arow = tid >> 1;             // 0..127
    const int acol = (tid & 1) << 4;       // 0 / 16
    const int kp   = tid >> 4;             // 0..15  (K pair index)
    const int bn   = (tid & 15) << 3;      // 0..120 (N chunk of 8)

    const size_t m0 = (size_t)blockIdx.y * 128;
    const int    n0 = blockIdx.x * 128;

    const unsigned short* Ab = A + (size_t)blockIdx.z * sA + m0 * (size_t)K
                                 + (size_t)arow * K + acol;
    const unsigned short* Bb = B + (size_t)blockIdx.z * sB + n0 + bn;

    v8f acc[2][4];
    #pragma unroll
    for (int r = 0; r < 2; ++r)
        #pragma unroll
        for (int c = 0; c < 4; ++c)
            #pragma unroll
            for (int e = 0; e < 8; ++e) acc[r][c][e] = 0.0f;

    uint4 a0, a1, b0, b1;
    auto gload = [&](int kb) {
        a0 = *(const uint4*)(Ab + kb);
        a1 = *(const uint4*)(Ab + kb + 8);
        b0 = *(const uint4*)(Bb + (size_t)(kb + 2 * kp    ) * N);
        b1 = *(const uint4*)(Bb + (size_t)(kb + 2 * kp + 1) * N);
    };
    auto sstore = [&](int p) {
        *(uint4*)(lA[p] + arow * LPA + acol)     = a0;
        *(uint4*)(lA[p] + arow * LPA + acol + 8) = a1;
        const unsigned short* p0 = (const unsigned short*)&b0;
        const unsigned short* p1 = (const unsigned short*)&b1;
        unsigned* lBu = (unsigned*)lB[p];
        #pragma unroll
        for (int i = 0; i < 8; ++i)
            lBu[(bn + i) * (LPB / 2) + kp] = (unsigned)p0[i] | ((unsigned)p1[i] << 16);
    };

    const int T = K >> 5;          // number of 32-wide K steps
    gload(0);
    sstore(0);
    __syncthreads();

    for (int i = 0; i < T; ++i) {
        const int p = i & 1;
        if (i + 1 < T) gload((i + 1) << 5);   // prefetch next tile (hidden by WMMAs)

        // A fragment per ISA 16-bit 16x32 layout:
        //   lane half 0: elems 0..7 = K 0..7,  elems 8..15 = K 16..23
        //   lane half 1: elems 0..7 = K 8..15, elems 8..15 = K 24..31
        Frag fa[2], fb[4];
        #pragma unroll
        for (int r = 0; r < 2; ++r) {
            const int base = (wm * 32 + r * 16 + l16) * LPA + lh * 8;
            fa[r].u[0] = *(const uint4*)(lA[p] + base);
            fa[r].u[1] = *(const uint4*)(lA[p] + base + 16);
        }
        // B fragment: lane = column, elems 0..7 = K lh*8..lh*8+7, elems 8..15 = K 16+lh*8..
        #pragma unroll
        for (int c = 0; c < 4; ++c) {
            const int base = (wn * 64 + c * 16 + l16) * LPB + lh * 8;
            fb[c].u[0] = *(const uint4*)(lB[p] + base);
            fb[c].u[1] = *(const uint4*)(lB[p] + base + 16);
        }
        #pragma unroll
        for (int r = 0; r < 2; ++r)
            #pragma unroll
            for (int c = 0; c < 4; ++c)
                acc[r][c] = __builtin_amdgcn_wmma_f32_16x16x32_bf16(
                    false, fa[r].v, false, fb[c].v,
                    (short)0, acc[r][c], false, false);

        if (i + 1 < T) {
            sstore(p ^ 1);          // other buffer: iter i-1 readers passed barrier
            __syncthreads();
        }
    }

    // epilogue: D layout = VGPR e -> M = e + 8*lh, N = lane&15
    #pragma unroll
    for (int r = 0; r < 2; ++r) {
        const int mrow = wm * 32 + r * 16 + lh * 8;
        #pragma unroll
        for (int c = 0; c < 4; ++c) {
            const int ncol = n0 + wn * 64 + c * 16 + l16;
            const size_t base = (size_t)blockIdx.z * sC +
                                (m0 + mrow) * (size_t)N + ncol;
            float bv = 0.0f;
            if (BIAS) bv = bias[ncol];
            #pragma unroll
            for (int e = 0; e < 8; ++e) {
                const size_t idx = base + (size_t)e * N;
                float v = acc[r][c][e];
                if (ACC) v += Cf[idx];
                v += bv;
                if      (ACT == 1) v = fmaxf(v, 0.0f);
                else if (ACT == 2) v = 0.5f * v * (1.0f + erff(v * 0.70710678118654752f));
                else if (ACT == 3) v = 1.0f / (1.0f + __expf(-v));
                if (WF) Cf[idx] = v;
                if (WB) Cb[idx] = f2bf(v);
            }
        }
    }
}

// ---------------------------------------------------------------------------
// Elementwise / reduction helpers
// ---------------------------------------------------------------------------
__global__ void k_f2bf(const float* __restrict__ s, unsigned short* __restrict__ d, int n) {
    int i = blockIdx.x * 256 + threadIdx.x;
    if (i < n) d[i] = f2bf(s[i]);
}

// fold 4H gate weight into two HxH bf16 matrices:
//   W1' = Wa + Wc + Wd ; W2' = Wb + Wc - Wd   (gate_in = [g1,g2,g1+g2,g1-g2])
__global__ void k_gatew(const float* __restrict__ gw,
                        unsigned short* __restrict__ w1,
                        unsigned short* __restrict__ w2) {
    int i = blockIdx.x * 256 + threadIdx.x;
    if (i >= 768 * 768) return;
    int r = i / 768, c = i % 768;
    float a  = gw[(size_t)(       r) * 768 + c];
    float b  = gw[(size_t)( 768 + r) * 768 + c];
    float cc = gw[(size_t)(1536 + r) * 768 + c];
    float dd = gw[(size_t)(2304 + r) * 768 + c];
    w1[i] = f2bf(a + cc + dd);
    w2[i] = f2bf(b + cc - dd);
}

// dinv[c][b][i] = rsqrt( sum_j graphs[b][c][i][j] )
__global__ __launch_bounds__(256)
void k_dinv(const float* __restrict__ graphs, float* __restrict__ dinv) {
    const int bid = blockIdx.x;                 // 2*8*1024 rows
    const int c = bid >> 13;
    const int b = (bid >> 10) & 7;
    const int i = bid & 1023;
    const float* row = graphs + (((size_t)b * 2 + c) * 1024 + i) * 1024;
    float s = 0.0f;
    for (int j = threadIdx.x; j < 1024; j += 256) s += row[j];
    __shared__ float red[256];
    red[threadIdx.x] = s; __syncthreads();
    for (int o = 128; o > 0; o >>= 1) {
        if (threadIdx.x < o) red[threadIdx.x] += red[threadIdx.x + o];
        __syncthreads();
    }
    if (threadIdx.x == 0) dinv[((size_t)c * 8 + b) * 1024 + i] = rsqrtf(red[0]);
}

// adj_bf[c][b][i][j] = bf16( graphs[b][c][i][j] * dinv_i * dinv_j )
__global__ void k_adj(const float* __restrict__ graphs, const float* __restrict__ dinv,
                      unsigned short* __restrict__ adj) {
    size_t idx = (size_t)blockIdx.x * 256 + threadIdx.x;  // 2*8*1024*1024
    int j = (int)(idx & 1023);
    int i = (int)((idx >> 10) & 1023);
    int b = (int)((idx >> 20) & 7);
    int c = (int)(idx >> 23);
    float v = graphs[(((size_t)b * 2 + c) * 1024 + i) * 1024 + j];
    const float* dv = dinv + ((size_t)c * 8 + b) * 1024;
    adj[idx] = f2bf(v * dv[i] * dv[j]);
}

// y = LN(nodes + dx) * g + b ; write f32 + bf16 copies.  One block per row (H=768).
__global__ __launch_bounds__(256)
void k_ln(const float* __restrict__ x0, const float* __restrict__ dx,
          const float* __restrict__ g,  const float* __restrict__ bb,
          float* __restrict__ of, unsigned short* __restrict__ ob) {
    const size_t row = blockIdx.x;
    const int t = threadIdx.x;
    const float* xr = x0 + row * 768;
    const float* dr = dx + row * 768;
    float v[3]; float s = 0.0f;
    #pragma unroll
    for (int i = 0; i < 3; ++i) { v[i] = xr[t + 256 * i] + dr[t + 256 * i]; s += v[i]; }
    __shared__ float red[256];
    red[t] = s; __syncthreads();
    for (int o = 128; o > 0; o >>= 1) { if (t < o) red[t] += red[t + o]; __syncthreads(); }
    const float mu = red[0] * (1.0f / 768.0f);
    __syncthreads();
    float s2 = 0.0f;
    #pragma unroll
    for (int i = 0; i < 3; ++i) { float d = v[i] - mu; s2 += d * d; }
    red[t] = s2; __syncthreads();
    for (int o = 128; o > 0; o >>= 1) { if (t < o) red[t] += red[t + o]; __syncthreads(); }
    const float rstd = rsqrtf(red[0] * (1.0f / 768.0f) + 1e-12f);
    #pragma unroll
    for (int i = 0; i < 3; ++i) {
        const int col = t + 256 * i;
        const float y = (v[i] - mu) * rstd * g[col] + bb[col];
        of[row * 768 + col] = y;
        ob[row * 768 + col] = f2bf(y);
    }
}

__global__ void k_mix(const float* __restrict__ gate, const float* __restrict__ g1,
                      const float* __restrict__ g2, float* __restrict__ out, int n) {
    int i = blockIdx.x * 256 + threadIdx.x;
    if (i < n) { float t = gate[i]; out[i] = t * g1[i] + (1.0f - t) * g2[i]; }
}

// ---------------------------------------------------------------------------
// Host driver
// ---------------------------------------------------------------------------
template<int ACT, bool ACC, bool WF, bool WB, bool BIAS>
static inline void gemm(hipStream_t st, const unsigned short* A, const unsigned short* B,
                        float* Cf, unsigned short* Cb, const float* bias,
                        int M, int N, int K, int batch,
                        long long sA, long long sB, long long sC) {
    dim3 grid(N / 128, M / 128, batch), blk(256);
    k_gemm<ACT, ACC, WF, WB, BIAS><<<grid, blk, 0, st>>>(A, B, Cf, Cb, bias,
                                                         M, N, K, sA, sB, sC);
}

extern "C" void kernel_launch(void* const* d_in, const int* in_sizes, int n_in,
                              void* d_out, int out_size, void* d_ws, size_t ws_size,
                              hipStream_t stream) {
    (void)in_sizes; (void)n_in; (void)out_size; (void)ws_size;
    const int Nn = 1024, Hh = 768, Ii = 3072, BN = 8 * 1024;

    const float* nodes  = (const float*)d_in[0];
    const float* graphs = (const float*)d_in[1];
    const float* wsrc[8] = {                 // fp32 weight matrices to convert
        (const float*)d_in[2], (const float*)d_in[3],   // gcn1 w1,w2
        (const float*)d_in[4], (const float*)d_in[5],   // gcn2 w1,w2
        (const float*)d_in[6], (const float*)d_in[8],   // exp1, col1
        (const float*)d_in[10], (const float*)d_in[12]  // exp2, col2
    };
    const float* exp_b[2] = { (const float*)d_in[7],  (const float*)d_in[11] };
    const float* col_b[2] = { (const float*)d_in[9],  (const float*)d_in[13] };
    const float* gate_w   = (const float*)d_in[14];
    const float* gate_b   = (const float*)d_in[15];
    const float* ln_g[2]  = { (const float*)d_in[16], (const float*)d_in[18] };
    const float* ln_b[2]  = { (const float*)d_in[17], (const float*)d_in[19] };

    // workspace carve-up (~340 MB)
    size_t off = 0;
    auto alloc = [&](size_t bytes) -> void* {
        void* p = (char*)d_ws + off;
        off += (bytes + 255) & ~(size_t)255;
        return p;
    };
    unsigned short* nodes_bf = (unsigned short*)alloc((size_t)BN * Hh * 2);
    unsigned short* wbf[8];
    for (int i = 0; i < 8; ++i) wbf[i] = (unsigned short*)alloc((size_t)Hh * Ii * 2);
    unsigned short* gw1  = (unsigned short*)alloc((size_t)Hh * Hh * 2);
    unsigned short* gw2  = (unsigned short*)alloc((size_t)Hh * Hh * 2);
    float*          dinv = (float*)alloc((size_t)2 * 8 * Nn * 4);
    unsigned short* adjb = (unsigned short*)alloc((size_t)2 * 8 * Nn * Nn * 2);
    unsigned short* t1   = (unsigned short*)alloc((size_t)BN * Ii * 2);
    unsigned short* t2   = (unsigned short*)alloc((size_t)BN * Ii * 2);
    unsigned short* gtmp = (unsigned short*)alloc((size_t)BN * Hh * 2);
    unsigned short* ggcn = (unsigned short*)alloc((size_t)BN * Hh * 2);
    float*          vf   = (float*)alloc((size_t)BN * Hh * 4);
    float*          gf[2]; unsigned short* gb[2];
    for (int i = 0; i < 2; ++i) {
        gf[i] = (float*)alloc((size_t)BN * Hh * 4);
        gb[i] = (unsigned short*)alloc((size_t)BN * Hh * 2);
    }
    float* gatef = (float*)alloc((size_t)BN * Hh * 4);

    // ---- prep: conversions, sym-normalize, gate weight folding ----
    {
        int n = BN * Hh;
        k_f2bf<<<(n + 255) / 256, 256, 0, stream>>>(nodes, nodes_bf, n);
        int nw = Hh * Ii;
        for (int i = 0; i < 8; ++i)
            k_f2bf<<<(nw + 255) / 256, 256, 0, stream>>>(wsrc[i], wbf[i], nw);
        k_gatew<<<(Hh * Hh + 255) / 256, 256, 0, stream>>>(gate_w, gw1, gw2);
        k_dinv<<<2 * 8 * Nn, 256, 0, stream>>>(graphs, dinv);
        k_adj<<<(int)(((size_t)2 * 8 * Nn * Nn) / 256), 256, 0, stream>>>(graphs, dinv, adjb);
    }

    const long long NN = (long long)Nn * Nn;
    const long long NI = (long long)Nn * Ii;
    const long long NH = (long long)Nn * Hh;

    // ---- per-branch: GCN (4 GEMMs) + FFN (2 GEMMs) + residual-LN ----
    for (int br = 0; br < 2; ++br) {
        const unsigned short* adjc = adjb + (size_t)br * 8 * Nn * Nn;
        const unsigned short* w1 = wbf[br * 2 + 0];
        const unsigned short* w2 = wbf[br * 2 + 1];
        const unsigned short* ew = wbf[4 + br * 2 + 0];
        const unsigned short* cw = wbf[4 + br * 2 + 1];

        // t1 = X @ W1                       [8192x3072] -> bf16
        gemm<0, false, false, true, false>(stream, nodes_bf, w1, nullptr, t1, nullptr,
                                           BN, Ii, Hh, 1, 0, 0, 0);
        // t2 = relu(adj @ t1)               batched 8 x [1024x3072] -> bf16
        gemm<1, false, false, true, false>(stream, adjc, t1, nullptr, t2, nullptr,
                                           Nn, Ii, Nn, 8, NN, NI, NI);
        // gtmp = t2 @ W2                    [8192x768] -> bf16
        gemm<0, false, false, true, false>(stream, t2, w2, nullptr, gtmp, nullptr,
                                           BN, Hh, Ii, 1, 0, 0, 0);
        // ggcn = relu(adj @ gtmp)           batched 8 x [1024x768] -> bf16
        gemm<1, false, false, true, false>(stream, adjc, gtmp, nullptr, ggcn, nullptr,
                                           Nn, Hh, Nn, 8, NN, NH, NH);
        // t1 = gelu(ggcn @ exp_w + exp_b)   [8192x3072] -> bf16
        gemm<2, false, false, true, true>(stream, ggcn, ew, nullptr, t1, exp_b[br],
                                          BN, Ii, Hh, 1, 0, 0, 0);
        // vf = t1 @ col_w + col_b           [8192x768] -> f32
        gemm<0, false, true, false, true>(stream, t1, cw, vf, nullptr, col_b[br],
                                          BN, Hh, Ii, 1, 0, 0, 0);
        // g = LN(nodes + vf)
        k_ln<<<BN, 256, 0, stream>>>(nodes, vf, ln_g[br], ln_b[br], gf[br], gb[br]);
    }

    // ---- gate: sigmoid(g1@W1' + g2@W2' + b), then mix ----
    gemm<0, false, true, false, false>(stream, gb[0], gw1, gatef, nullptr, nullptr,
                                       BN, Hh, Hh, 1, 0, 0, 0);
    gemm<3, true, true, false, true>(stream, gb[1], gw2, gatef, nullptr, gate_b,
                                     BN, Hh, Hh, 1, 0, 0, 0);
    k_mix<<<(BN * Hh + 255) / 256, 256, 0, stream>>>(gatef, gf[0], gf[1],
                                                     (float*)d_out, BN * Hh);
}
